// IntraModalityContrastLoss_11132555231827
// MI455X (gfx1250) — matrix operand order repaired and checked by reading, not compiled, not run on hardware
//
#include <hip/hip_runtime.h>

// NT-Xent / intra-modality contrastive loss, fused for gfx1250 (MI455X).
// sim = z z^T never materialized; exp-rowsum fused into the WMMA epilogue.
// B slabs staged through LDS (double-buffered), shared by 8 waves/WG;
// each wave holds TWO register-resident A slabs so every LDS B-fragment
// feeds two independent WMMA chains (halves DS traffic per WMMA).

typedef _Float16 v8h  __attribute__((ext_vector_type(8)));
typedef _Float16 v16h __attribute__((ext_vector_type(16)));
typedef float    v8f  __attribute__((ext_vector_type(8)));

#define D        256          // feature dim
#define NROWS    4096         // 2N
#define HALFN    2048         // N = b*m
#define NSPLIT   8            // column splits in simsum
#define INV_T    2.0f         // 1 / TEMP
#define EPSN     1e-12f

#define STAGEC   32           // columns staged per LDS buffer
#define NSTAGE   16           // 512-column span / 32
#define PADH     272          // padded halves per staged column (544 B: 8-bank lane shift)

// ---------------------------------------------------------------------------
// Kernel 1: row-normalize x_i (rows 0..2047) and x_j (rows 2048..4095) -> f16 z
// ---------------------------------------------------------------------------
__global__ void nt_normalize(const float* __restrict__ xi,
                             const float* __restrict__ xj,
                             _Float16* __restrict__ z)
{
    const int wave = threadIdx.x >> 5;
    const int lane = threadIdx.x & 31;
    const int row  = blockIdx.x * 8 + wave;          // 0..4095

    const float* __restrict__ src =
        (row < HALFN) ? (xi + (size_t)row * D) : (xj + (size_t)(row - HALFN) * D);

    const float4* s4 = (const float4*)src;
    float4 a = s4[lane * 2 + 0];
    float4 b = s4[lane * 2 + 1];

    float ss = a.x*a.x + a.y*a.y + a.z*a.z + a.w*a.w
             + b.x*b.x + b.y*b.y + b.z*b.z + b.w*b.w;
    ss += __shfl_xor(ss, 16);
    ss += __shfl_xor(ss, 8);
    ss += __shfl_xor(ss, 4);
    ss += __shfl_xor(ss, 2);
    ss += __shfl_xor(ss, 1);

    const float scale = 1.0f / fmaxf(sqrtf(ss), EPSN);

    v8h o;
    o[0] = (_Float16)(a.x * scale);
    o[1] = (_Float16)(a.y * scale);
    o[2] = (_Float16)(a.z * scale);
    o[3] = (_Float16)(a.w * scale);
    o[4] = (_Float16)(b.x * scale);
    o[5] = (_Float16)(b.y * scale);
    o[6] = (_Float16)(b.z * scale);
    o[7] = (_Float16)(b.w * scale);
    *(v8h*)(z + (size_t)row * D + lane * 8) = o;
}

// ---------------------------------------------------------------------------
// Load one 16-row x K=256 A slab (8 k-tiles of 16x32) for WMMA.
// k order used here (contiguous, same as B): elems 0..7 = K kb*32 + khalf*16 + (0..7),
//                        elems 8..15 = K kb*32 + khalf*16 + 8 + (0..7)
// ---------------------------------------------------------------------------
__device__ __forceinline__ void load_a_slab(const _Float16* __restrict__ arow,
                                            int khalf, v16h* A)
{
#pragma unroll
    for (int kb = 0; kb < 8; ++kb) {
        v8h lo = *(const v8h*)(arow + kb * 32 + khalf * 16);
        v8h hi = *(const v8h*)(arow + kb * 32 + khalf * 16 + 8);
        v16h af;
#pragma unroll
        for (int e = 0; e < 8; ++e) { af[e] = lo[e]; af[e + 8] = hi[e]; }
        A[kb] = af;
    }
}

// ---------------------------------------------------------------------------
// Kernel 2: fused GEMM + exp row-sum.
// grid = (16 rowgroups, 8 colsplits), block = 256 (8 waves).
// Wave w owns 32 rows: [rowgroup*256 + w*32, +32) as two A slabs.
// Per stage: 32 B-columns in LDS -> 2 subtiles x 8 k-steps x 2 WMMA per wave.
// ---------------------------------------------------------------------------
__global__ __launch_bounds__(256)
void nt_simsum(const _Float16* __restrict__ z,
               float* __restrict__ P)
{
    __shared__ _Float16 Bs[2][STAGEC][PADH];

    const int t        = threadIdx.x;
    const int lane     = t & 31;
    const int wave     = t >> 5;
    const int rowbase  = blockIdx.x * 256 + wave * 32;   // 32 rows per wave
    const int csplit   = blockIdx.y;
    const int colstart = csplit * 512;
    const int m        = lane & 15;
    const int khalf    = lane >> 4;

    // ---- two register-resident A slabs (rows +0..15 and +16..31) ----
    v16h A0[8], A1[8];
    load_a_slab(z + (size_t)(rowbase +      m) * D, khalf, A0);
    load_a_slab(z + (size_t)(rowbase + 16 + m) * D, khalf, A1);

    // ---- cooperative stage loader: thread t -> column t/8, halves (t%8)*32 ----
    const int lc = t >> 3;          // 0..31: column within stage
    const int lo = (t & 7) * 32;    // half offset within the 256-half column row

    const _Float16* __restrict__ g0 =
        z + (size_t)(colstart + lc) * D + lo;            // stage s: g0 + s*32*D

    // prologue: stage 0 -> LDS buffer 0
    {
        v16h r0 = *(const v16h*)(g0);
        v16h r1 = *(const v16h*)(g0 + 16);
        *(v16h*)(&Bs[0][lc][lo])      = r0;
        *(v16h*)(&Bs[0][lc][lo + 16]) = r1;
    }
    __syncthreads();

    v8f rs0 = {}, rs1 = {};                              // per-lane exp row-sums

    for (int s = 0; s < NSTAGE; ++s) {
        const int cur = s & 1;

        // issue next stage's global loads before compute (latency overlap)
        v16h r0, r1;
        if (s + 1 < NSTAGE) {
            const _Float16* g = g0 + (size_t)(s + 1) * STAGEC * D;
            r0 = *(const v16h*)(g);
            r1 = *(const v16h*)(g + 16);
        }

        // compute: 2 subtiles of 16 columns; each B fragment feeds 2 WMMAs
#pragma unroll
        for (int sub = 0; sub < 2; ++sub) {
            const _Float16* bp = &Bs[cur][sub * 16 + m][khalf * 16];
            v8f c0 = {}, c1 = {};
#pragma unroll
            for (int kb = 0; kb < 8; ++kb) {
                v16h bf = *(const v16h*)(bp + kb * 32);
                c0 = __builtin_amdgcn_wmma_f32_16x16x32_f16(
                         false, A0[kb], false, bf, (short)0, c0, false, false);
                c1 = __builtin_amdgcn_wmma_f32_16x16x32_f16(
                         false, A1[kb], false, bf, (short)0, c1, false, false);
            }
#pragma unroll
            for (int r = 0; r < 8; ++r) {
                rs0[r] += __expf(c0[r] * INV_T);
                rs1[r] += __expf(c1[r] * INV_T);
            }
        }

        __syncthreads();                 // everyone done reading Bs[cur]
        if (s + 1 < NSTAGE) {
            *(v16h*)(&Bs[cur ^ 1][lc][lo])      = r0;
            *(v16h*)(&Bs[cur ^ 1][lc][lo + 16]) = r1;
        }
        __syncthreads();                 // next buffer fully populated
    }

    // ---- reduce across the 16 lanes of each half (masks <= 8 stay in-half) ----
#pragma unroll
    for (int r = 0; r < 8; ++r) {
        float v0 = rs0[r];
        v0 += __shfl_xor(v0, 1);
        v0 += __shfl_xor(v0, 2);
        v0 += __shfl_xor(v0, 4);
        v0 += __shfl_xor(v0, 8);
        rs0[r] = v0;
        float v1 = rs1[r];
        v1 += __shfl_xor(v1, 1);
        v1 += __shfl_xor(v1, 2);
        v1 += __shfl_xor(v1, 4);
        v1 += __shfl_xor(v1, 8);
        rs1[r] = v1;
    }

    if ((lane & 15) == 0) {
        const int rb = rowbase + khalf * 8;  // lane 0 -> M=r, lane 16 -> M=r+8
#pragma unroll
        for (int r = 0; r < 8; ++r) {
            P[(size_t)(rb + r)      * NSPLIT + csplit] = rs0[r];
            P[(size_t)(rb + r + 16) * NSPLIT + csplit] = rs1[r];
        }
    }
}

// ---------------------------------------------------------------------------
// Kernel 3: final reduction. Single block, 256 threads, fixed-order sums.
// loss = (1/2N) * sum_i [ log(rowsum_i - exp(self_i/T)) - pos_i/T ]
// ---------------------------------------------------------------------------
__global__ void nt_loss(const _Float16* __restrict__ z,
                        const float* __restrict__ P,
                        float* __restrict__ out)
{
    __shared__ float red[256];
    const int t = threadIdx.x;
    float acc = 0.0f;

    for (int row = t; row < NROWS; row += 256) {
        float rowsum = 0.0f;
#pragma unroll
        for (int s = 0; s < NSPLIT; ++s)
            rowsum += P[(size_t)row * NSPLIT + s];

        const int pair = (row < HALFN) ? row + HALFN : row - HALFN;
        const _Float16* zr = z + (size_t)row  * D;
        const _Float16* zp = z + (size_t)pair * D;

        float self = 0.0f, pos = 0.0f;
#pragma unroll 8
        for (int k = 0; k < D; ++k) {
            float a = (float)zr[k];
            float b = (float)zp[k];
            self = fmaf(a, a, self);
            pos  = fmaf(a, b, pos);
        }

        const float denom = rowsum - __expf(self * INV_T);
        acc += __logf(denom) - pos * INV_T;
    }

    red[t] = acc;
    __syncthreads();
#pragma unroll
    for (int s = 128; s > 0; s >>= 1) {
        if (t < s) red[t] += red[t + s];
        __syncthreads();
    }
    if (t == 0) out[0] = red[0] / (float)NROWS;
}

// ---------------------------------------------------------------------------
extern "C" void kernel_launch(void* const* d_in, const int* in_sizes, int n_in,
                              void* d_out, int out_size, void* d_ws, size_t ws_size,
                              hipStream_t stream)
{
    (void)in_sizes; (void)n_in; (void)out_size; (void)ws_size;

    const float* xi = (const float*)d_in[0];   // (512,4,256) f32 -> 2048x256
    const float* xj = (const float*)d_in[1];   // (512,4,256) f32 -> 2048x256

    const size_t Z_BYTES = (size_t)NROWS * D * sizeof(_Float16);   // 2 MiB
    _Float16* z16 = (_Float16*)d_ws;
    float*    P   = (float*)((char*)d_ws + Z_BYTES);               // 4096*8 f32

    // 1) normalize -> f16 z  (512 blocks x 8 waves)
    nt_normalize<<<dim3(NROWS / 8), dim3(256), 0, stream>>>(xi, xj, z16);

    // 2) fused WMMA GEMM + exp row-sums (16 rowgroups x 8 column splits)
    nt_simsum<<<dim3(NROWS / 256, NSPLIT), dim3(256), 0, stream>>>(z16, P);

    // 3) scalar loss
    nt_loss<<<dim3(1), dim3(256), 0, stream>>>(z16, P, (float*)d_out);
}
